// GIBLiKPResidualBlock_6837587935397
// MI455X (gfx1250) — matrix operand order repaired and checked by reading, not compiled, not run on hardware
//
#include <hip/hip_runtime.h>

#define N_PTS 32768
#define KNBR  32
#define PKP   15
#define CIN   128
#define COUT  256
#define MID   64
#define NOBS  16
#define GN_EPS 1e-5f
#define SLOPE  0.1f

typedef __attribute__((ext_vector_type(16))) __bf16 v16bf;
typedef __attribute__((ext_vector_type(8)))  __bf16 v8bf;
typedef __attribute__((ext_vector_type(8)))  float  v8f;
typedef unsigned int u32x4 __attribute__((ext_vector_type(4)));
typedef int          i32x8 __attribute__((ext_vector_type(8)));
typedef int          i32x4 __attribute__((ext_vector_type(4)));

static __device__ __forceinline__ v8f wmma_bf16(v16bf a, v16bf b, v8f c) {
  // D(16x16,f32) = A(16x32,bf16) * B(32x16,bf16) + C
  return __builtin_amdgcn_wmma_f32_16x16x32_bf16(false, a, false, b, (short)0, c,
                                                 false, false);
}

static __device__ __forceinline__ v16bf bfcat(v8bf a, v8bf b) {
  return __builtin_shufflevector(a, b, 0,1,2,3,4,5,6,7,8,9,10,11,12,13,14,15);
}

// A-matrix (16x32 bf16) fragment from an fp32 row-major row.
// Per ISA: lane l holds row m=l&15, K = kb..kb+7 and kb+16..kb+23, kb=(l>>4)*8.
static __device__ __forceinline__ v16bf load_a_f32(const float* rowbase, int col) {
  v8f f0 = *(const v8f*)(rowbase + col);
  v8f f1 = *(const v8f*)(rowbase + col + 16);
  v16bf a;
#pragma unroll
  for (int j = 0; j < 8; ++j) { a[j] = (__bf16)f0[j]; a[8 + j] = (__bf16)f1[j]; }
  return a;
}

static __device__ __forceinline__ float rsum8(float v) {
  v += __shfl_xor(v, 1, 32);
  v += __shfl_xor(v, 2, 32);
  v += __shfl_xor(v, 4, 32);
  return v;
}
static __device__ __forceinline__ float rsum16(float v) {
  v = rsum8(v);
  v += __shfl_xor(v, 8, 32);
  return v;
}

// Truncate a flat pointer to its LDS byte offset (LDS aperture keeps the
// wave-relative offset in the low 32 bits).
static __device__ __forceinline__ unsigned lds_off(const void* p) {
  return (unsigned)(unsigned long long)p;
}

// ---------------------------------------------------------------------------
// Weight prep: fp32 [K x ncols] row-major -> bf16 WMMA B-tiles (32x16 each).
// Tile (kt,nt): lane l holds N = nt*16+(l&15), K = kt*32+(l>>4)*8 + j (+8 for j>=8), j=0..15.
// ---------------------------------------------------------------------------
__global__ void prep_weights(const float* __restrict__ src, __bf16* __restrict__ dst,
                             int KT, int NT, int Kreal, int ncols) {
  int t = blockIdx.x * blockDim.x + threadIdx.x;
  int lane = t & 31;
  int tile = t >> 5;
  if (tile >= KT * NT) return;
  int kt = tile / NT, nt = tile % NT;
  int kbase = kt * 32 + (lane >> 4) * 8;
  int c = nt * 16 + (lane & 15);
  __bf16* out = dst + (size_t)tile * 512 + lane * 16;
#pragma unroll
  for (int j = 0; j < 16; ++j) {
    int k = kbase + j + (j >= 8 ? 8 : 0);
    float v = (k < Kreal) ? src[(size_t)k * ncols + c] : 0.0f;
    out[j] = (__bf16)v;
  }
}

// ---------------------------------------------------------------------------
// K1: residual = ([s_feats@W1 | exp(-50*d2(obs))] @ Wproj)  -> bf16 [N x 64]
// 4 waves/block, each wave owns 16 points. Weight tiles (28 KB, reused by every
// block) are DMA'd into LDS once per workgroup via the Tensor Data Mover.
// ---------------------------------------------------------------------------
__global__ __launch_bounds__(128) void k1_residual(
    const float* __restrict__ s_points, const float* __restrict__ s_feats,
    const float* __restrict__ obs_centers,
    const __bf16* __restrict__ wt_pack,  // wt_u1 (16 tiles) ++ wt_proj (12 tiles)
    __bf16* __restrict__ resid_out) {
  __shared__ __align__(16) __bf16 wt_lds[14336];  // 28672 B
  __shared__ __align__(16) __bf16 A2[4][16 * 96];
  int lane = threadIdx.x & 31;
  int w = threadIdx.x >> 5;
  int m16 = lane & 15;
  int hi = lane >> 4;
  int kb = hi * 8;
  int pt_base = blockIdx.x * 64 + w * 16;

  if (w == 0) {
    // TDM: one descriptor, 1-row tile of 7168 dwords, global -> LDS.
    unsigned long long ga = (unsigned long long)wt_pack;
    u32x4 g0;
    g0.x = 1u;                                          // count=1, user mode
    g0.y = lds_off(&wt_lds[0]);                         // lds_addr
    g0.z = (unsigned)(ga & 0xFFFFFFFFu);                // global_addr[31:0]
    g0.w = (unsigned)((ga >> 32) & 0x01FFFFFFu)         // global_addr[56:32]
           | (2u << 30);                                // type=2 ("image")
    i32x8 g1;
    g1[0] = 0x00020000;                  // wg_mask=0, data_size=4B
    g1[1] = (int)((7168u & 0xFFFFu) << 16);  // tensor_dim0[15:0]=7168
    g1[2] = (int)(1u << 16);             // tensor_dim0 hi=0, tensor_dim1=1
    g1[3] = (int)((7168u & 0xFFFFu) << 16);  // tile_dim0=7168
    g1[4] = 1;                           // tile_dim1=1, tile_dim2=0
    g1[5] = 7168;                        // tensor_dim0_stride
    g1[6] = 0;
    g1[7] = 0;
    i32x4 gz = {0, 0, 0, 0};
    i32x8 gz8 = {0, 0, 0, 0, 0, 0, 0, 0};
    __builtin_amdgcn_tensor_load_to_lds(g0, g1, gz, gz, gz8, 0);
    __builtin_amdgcn_s_wait_tensorcnt(0);
  }
  __syncthreads();

  // h = s_feats[16x128] @ W1[128x64]   (B tiles from LDS)
  v8f acc[4] = {};
  const float* arow = s_feats + (size_t)(pt_base + m16) * CIN;
#pragma unroll
  for (int kt = 0; kt < 4; ++kt) {
    v16bf a = load_a_f32(arow, kt * 32 + kb);
#pragma unroll
    for (int j = 0; j < 4; ++j) {
      v16bf b = *(const v16bf*)(wt_lds + (size_t)(kt * 4 + j) * 512 + lane * 16);
      acc[j] = wmma_bf16(a, b, acc[j]);
    }
  }
  // stage h into LDS A-tile rows [16 x 96] (k 0..63)
  __bf16* A2w = &A2[w][0];
#pragma unroll
  for (int j = 0; j < 4; ++j)
#pragma unroll
    for (int r = 0; r < 8; ++r) {
      int row = r + hi * 8;
      A2w[row * 96 + j * 16 + m16] = (__bf16)acc[j][r];
    }
  // observer features (k 64..79) + zero pad (k 80..95)
  {
    int row = m16;
    int pt = pt_base + row;
    float px = s_points[3 * pt + 0], py = s_points[3 * pt + 1], pz = s_points[3 * pt + 2];
#pragma unroll
    for (int jj = 0; jj < 8; ++jj) {
      int o = hi * 8 + jj;
      float dx = px - obs_centers[3 * o + 0];
      float dy = py - obs_centers[3 * o + 1];
      float dz = pz - obs_centers[3 * o + 2];
      float d2 = dx * dx + dy * dy + dz * dz;
      A2w[row * 96 + 64 + o] = (__bf16)__expf(-50.0f * d2);  // 1/(2*0.1^2)=50
      A2w[row * 96 + 80 + o] = (__bf16)0.0f;
    }
  }
  __syncthreads();
  // g = A2[16x96] @ Wproj_pad[96x64]  (proj tiles start at LDS element 8192)
  v8f acc2[4] = {};
#pragma unroll
  for (int kt = 0; kt < 3; ++kt) {
    int base = m16 * 96 + kt * 32 + kb;
    v16bf a = bfcat(*(const v8bf*)(A2w + base), *(const v8bf*)(A2w + base + 16));
#pragma unroll
    for (int j = 0; j < 4; ++j) {
      v16bf b = *(const v16bf*)(wt_lds + 8192 + (size_t)(kt * 4 + j) * 512 + lane * 16);
      acc2[j] = wmma_bf16(a, b, acc2[j]);
    }
  }
#pragma unroll
  for (int j = 0; j < 4; ++j)
#pragma unroll
    for (int r = 0; r < 8; ++r) {
      int row = r + hi * 8;
      resid_out[(size_t)(pt_base + row) * MID + j * 16 + m16] = (__bf16)acc2[j][r];
    }
}

// ---------------------------------------------------------------------------
// K2: KPConv. Block = 16 points, 4 waves.
// Phase 1 (per point): async-gather nb feats into LDS, influences,
//                      agg = infl^T(16x32) x nbf(32x64).
// Phase 2: conv[16x64] = agg[16x960] @ Wkp[960x64]; GN1(8ch) + leaky -> bf16.
// ---------------------------------------------------------------------------
__global__ __launch_bounds__(128) void k2_kpconv(
    const float* __restrict__ q_points, const float* __restrict__ s_points,
    const int* __restrict__ nbr, const float* __restrict__ kpts,
    const __bf16* __restrict__ resid, const __bf16* __restrict__ wt_kp,
    const float* __restrict__ gn1_scale, const float* __restrict__ gn1_bias,
    __bf16* __restrict__ conv_out) {
  __shared__ __align__(16) __bf16 nbf[4][KNBR * MID];   // 4 x 4KB
  __shared__ __align__(16) __bf16 infl[4][KNBR * 16];   // 4 x 1KB
  __shared__ __align__(16) __bf16 agg[16 * 968];        // 16 pts x 960 (+pad)
  int lane = threadIdx.x & 31;
  int w = threadIdx.x >> 5;
  int m16 = lane & 15;
  int hi = lane >> 4;
  int kb = hi * 8;

  __bf16* nbfW = &nbf[w][0];
  __bf16* inflW = &infl[w][0];

  for (int t = 0; t < 4; ++t) {
    int pg = w * 4 + t;
    int pt = blockIdx.x * 16 + pg;
    int nb = nbr[pt * KNBR + lane];  // lane = neighbor index k
    // async gather: neighbor feature row (128 B) global -> LDS, no VGPR round-trip
    {
      unsigned la = lds_off(nbfW + (size_t)lane * MID);
      unsigned long long ga = (unsigned long long)(resid + (size_t)nb * MID);
#pragma unroll
      for (int j = 0; j < 8; ++j) {
        unsigned laj = la + j * 16;
        unsigned long long gaj = ga + (unsigned long long)(j * 16);
        asm volatile("global_load_async_to_lds_b128 %0, %1, off"
                     :: "v"(laj), "v"(gaj) : "memory");
      }
    }
    float qx = q_points[3 * pt + 0], qy = q_points[3 * pt + 1], qz = q_points[3 * pt + 2];
    float dx = s_points[3 * nb + 0] - qx;
    float dy = s_points[3 * nb + 1] - qy;
    float dz = s_points[3 * nb + 2] - qz;
#pragma unroll
    for (int p = 0; p < PKP; ++p) {
      float ex = dx - kpts[3 * p + 0];
      float ey = dy - kpts[3 * p + 1];
      float ez = dz - kpts[3 * p + 2];
      float r = sqrtf(ex * ex + ey * ey + ez * ez);
      inflW[lane * 16 + p] = (__bf16)fmaxf(0.0f, 1.0f - r * 10.0f);  // 1/SIGMA=10
    }
    inflW[lane * 16 + 15] = (__bf16)0.0f;
    asm volatile("s_wait_asynccnt 0x0" ::: "memory");
    __syncthreads();
    // A = infl^T : 16(P) x 32(K)
    v16bf a;
#pragma unroll
    for (int j = 0; j < 8; ++j) {
      a[j]     = inflW[(kb + j) * 16 + m16];
      a[8 + j] = inflW[(kb + 16 + j) * 16 + m16];
    }
    int kb2 = hi * 8;
    v8f acc[4] = {};
#pragma unroll
    for (int j = 0; j < 4; ++j) {
      int c = j * 16 + m16;
      v16bf b;
#pragma unroll
      for (int e = 0; e < 16; ++e) b[e] = nbfW[(kb2 + e + (e >= 8 ? 8 : 0)) * MID + c];
      acc[j] = wmma_bf16(a, b, acc[j]);
    }
    // agg row for this point, flattened k = p*64 + c
#pragma unroll
    for (int j = 0; j < 4; ++j)
#pragma unroll
      for (int r = 0; r < 8; ++r) {
        int p = r + hi * 8;
        if (p < PKP) agg[pg * 968 + p * MID + j * 16 + m16] = (__bf16)acc[j][r];
      }
    __syncthreads();
  }

  // Phase 2: wave w owns N-subtile nt = w
  v8f acc2 = {};
  for (int kt = 0; kt < 30; ++kt) {
    if (kt + 1 < 30)
      __builtin_prefetch(wt_kp + (size_t)((kt + 1) * 4 + w) * 512 + lane * 16, 0, 1);
    int base = m16 * 968 + kt * 32 + kb;
    v16bf a = bfcat(*(const v8bf*)(agg + base), *(const v8bf*)(agg + base + 16));
    v16bf b = *(const v16bf*)(wt_kp + (size_t)(kt * 4 + w) * 512 + lane * 16);
    acc2 = wmma_bf16(a, b, acc2);
  }
  // GN1 (groups of 8 channels) + leaky, store bf16 [N x 64]
  int c = w * 16 + m16;
  float sc = gn1_scale[c], bi = gn1_bias[c];
#pragma unroll
  for (int r = 0; r < 8; ++r) {
    float x = acc2[r];
    float s = rsum8(x);
    float ss = rsum8(x * x);
    float mu = s * 0.125f;
    float var = ss * 0.125f - mu * mu;
    float y = (x - mu) * rsqrtf(var + GN_EPS) * sc + bi;
    y = y >= 0.0f ? y : SLOPE * y;
    int m = r + hi * 8;
    conv_out[(size_t)(blockIdx.x * 16 + m) * MID + c] = (__bf16)y;
  }
}

// GroupNorm over 32-channel groups fully contained in one wave's 64-ch slice.
static __device__ __forceinline__ void groupnorm32(v8f acc[4], const float* scale,
                                                   const float* bias, int s, int m16) {
#pragma unroll
  for (int g = 0; g < 2; ++g) {
    int j0 = 2 * g, j1 = 2 * g + 1;
#pragma unroll
    for (int r = 0; r < 8; ++r) {
      float x0 = acc[j0][r], x1 = acc[j1][r];
      float sum = rsum16(x0) + rsum16(x1);
      float ss = rsum16(x0 * x0) + rsum16(x1 * x1);
      float mu = sum * (1.0f / 32.0f);
      float var = ss * (1.0f / 32.0f) - mu * mu;
      float inv = rsqrtf(var + GN_EPS);
      int c0 = s * 64 + j0 * 16 + m16;
      int c1 = s * 64 + j1 * 16 + m16;
      acc[j0][r] = (x0 - mu) * inv * scale[c0] + bias[c0];
      acc[j1][r] = (x1 - mu) * inv * scale[c1] + bias[c1];
    }
  }
}

// ---------------------------------------------------------------------------
// K3: out = leaky(GN2(conv@W2) + GNs(s_feats@Wshort)).
// Block = 16 points, wave w owns channel slice [64w, 64w+64).
// ---------------------------------------------------------------------------
__global__ __launch_bounds__(128) void k3_output(
    const float* __restrict__ s_feats, const __bf16* __restrict__ conv_in,
    const __bf16* __restrict__ wt_u2, const __bf16* __restrict__ wt_short,
    const float* __restrict__ gn2_scale, const float* __restrict__ gn2_bias,
    const float* __restrict__ gns_scale, const float* __restrict__ gns_bias,
    float* __restrict__ out) {
  int lane = threadIdx.x & 31;
  int s = threadIdx.x >> 5;
  int m16 = lane & 15;
  int hi = lane >> 4;
  int kb = hi * 8;
  int pt_base = blockIdx.x * 16;

  // res = conv[16x64] @ W2[:, 64s..64s+63]
  v8f accR[4] = {};
  const __bf16* crow = conv_in + (size_t)(pt_base + m16) * MID;
#pragma unroll
  for (int kt = 0; kt < 2; ++kt) {
    v16bf a = bfcat(*(const v8bf*)(crow + kt * 32 + kb),
                    *(const v8bf*)(crow + kt * 32 + kb + 16));
#pragma unroll
    for (int j = 0; j < 4; ++j) {
      v16bf b = *(const v16bf*)(wt_u2 + (size_t)(kt * 16 + s * 4 + j) * 512 + lane * 16);
      accR[j] = wmma_bf16(a, b, accR[j]);
    }
  }
  groupnorm32(accR, gn2_scale, gn2_bias, s, m16);

  // short = s_feats[16x128] @ Wshort[:, 64s..64s+63]
  v8f accS[4] = {};
  const float* frow = s_feats + (size_t)(pt_base + m16) * CIN;
#pragma unroll
  for (int kt = 0; kt < 4; ++kt) {
    v16bf a = load_a_f32(frow, kt * 32 + kb);
#pragma unroll
    for (int j = 0; j < 4; ++j) {
      v16bf b = *(const v16bf*)(wt_short + (size_t)(kt * 16 + s * 4 + j) * 512 + lane * 16);
      accS[j] = wmma_bf16(a, b, accS[j]);
    }
  }
  groupnorm32(accS, gns_scale, gns_bias, s, m16);

#pragma unroll
  for (int j = 0; j < 4; ++j)
#pragma unroll
    for (int r = 0; r < 8; ++r) {
      float y = accR[j][r] + accS[j][r];
      y = y >= 0.0f ? y : SLOPE * y;
      int m = r + hi * 8;
      out[(size_t)(pt_base + m) * COUT + s * 64 + j * 16 + m16] = y;
    }
}

// ---------------------------------------------------------------------------
extern "C" void kernel_launch(void* const* d_in, const int* in_sizes, int n_in,
                              void* d_out, int out_size, void* d_ws, size_t ws_size,
                              hipStream_t stream) {
  const float* q_points = (const float*)d_in[0];
  const float* s_points = (const float*)d_in[1];
  const float* s_feats  = (const float*)d_in[2];
  const int*   nbr      = (const int*)d_in[3];
  // d_in[4]: lengths (equal, unused)
  const float* w_u1  = (const float*)d_in[5];
  const float* obs_c = (const float*)d_in[6];
  const float* w_pr  = (const float*)d_in[7];
  const float* kpts  = (const float*)d_in[8];
  const float* kp_w  = (const float*)d_in[9];
  const float* gn1s  = (const float*)d_in[10];
  const float* gn1b  = (const float*)d_in[11];
  const float* w_u2  = (const float*)d_in[12];
  const float* gn2s  = (const float*)d_in[13];
  const float* gn2b  = (const float*)d_in[14];
  const float* w_sh  = (const float*)d_in[15];
  const float* gnss  = (const float*)d_in[16];
  const float* gnsb  = (const float*)d_in[17];

  char* ws = (char*)d_ws;
  __bf16* wt_u1 = (__bf16*)(ws + 0);          // 16 tiles * 1KB  (packed with proj)
  __bf16* wt_pr = (__bf16*)(ws + 16384);      // 12 tiles
  __bf16* wt_kp = (__bf16*)(ws + 28672);      // 120 tiles
  __bf16* wt_u2 = (__bf16*)(ws + 151552);     // 32 tiles
  __bf16* wt_sh = (__bf16*)(ws + 184320);     // 64 tiles
  __bf16* resid = (__bf16*)(ws + 249856);     // N x 64 bf16
  __bf16* convb = (__bf16*)(ws + 249856 + (size_t)N_PTS * MID * 2);

  auto prep = [&](const float* src, __bf16* dst, int KT, int NT, int Kreal, int nc) {
    int threads = KT * NT * 32;
    prep_weights<<<(threads + 255) / 256, 256, 0, stream>>>(src, dst, KT, NT, Kreal, nc);
  };
  prep(w_u1, wt_u1, 4, 4, 128, 64);
  prep(w_pr, wt_pr, 3, 4, 80, 64);      // K padded 80 -> 96
  prep(kp_w, wt_kp, 30, 4, 960, 64);    // [15,64,64] flattened to [960,64]
  prep(w_u2, wt_u2, 2, 16, 64, 256);
  prep(w_sh, wt_sh, 4, 16, 128, 256);

  k1_residual<<<N_PTS / 64, 128, 0, stream>>>(s_points, s_feats, obs_c, wt_u1, resid);
  k2_kpconv<<<N_PTS / 16, 128, 0, stream>>>(q_points, s_points, nbr, kpts, resid, wt_kp,
                                            gn1s, gn1b, convb);
  k3_output<<<N_PTS / 16, 128, 0, stream>>>(s_feats, convb, wt_u2, wt_sh, gn2s, gn2b,
                                            gnss, gnsb, (float*)d_out);
}
